// FastMultiHeadAttention_67602785239407
// MI455X (gfx1250) — matrix operand order repaired and checked by reading, not compiled, not run on hardware
//
#include <hip/hip_runtime.h>
#include <hip/hip_bf16.h>
#include <stdint.h>

// ---------------------------------------------------------------------------
// FastMultiHeadAttention for MI455X (gfx1250): bf16 WMMA pipeline with
// double/triple-buffered Tensor-Data-Mover (TDM) prefetch into LDS.
//   1) QKV GEMM  (fp32 -> bf16 WMMA; A tile + B panel TDM-prefetched)
//   2) RoPE + Q/K/V split (bf16, [B,S,H,D] layout)
//   3) Causal flash attention: S^T = K Q^T, O^T = V^T P^T; K/V chunks
//      TDM-prefetched (triple buffer) while WMMAs run on the current chunk.
//   4) Output projection GEMM (bf16 WMMA, f32 out)
// ---------------------------------------------------------------------------

typedef __bf16 bf;
typedef __attribute__((ext_vector_type(16))) __bf16 v16bf;
typedef __attribute__((ext_vector_type(8)))  float  v8f;
typedef __attribute__((ext_vector_type(4)))  unsigned u32x4;
typedef __attribute__((ext_vector_type(8)))  int      i32x8;
typedef __attribute__((ext_vector_type(4)))  int      i32x4;

#define B_SZ   4
#define S_SZ   2048
#define HID_SZ 2048
#define NH_SZ  16
#define NKV_SZ 4
#define HD_SZ  128
#define QKV_N  3072           // (NH + 2*NKV) * HD
#define ATT_SCALE 0.08838834764831845f   // HD^-0.5

#if __has_builtin(__builtin_amdgcn_tensor_load_to_lds) && \
    __has_builtin(__builtin_amdgcn_s_wait_tensorcnt)
#define USE_TDM 1
#else
#define USE_TDM 0
#endif

__device__ __forceinline__ bf f2bf(float f) {
  unsigned u = __builtin_bit_cast(unsigned, f);
  unsigned r = u + 0x7FFFu + ((u >> 16) & 1u);   // round-to-nearest-even
  unsigned short h = (unsigned short)(r >> 16);
  return __builtin_bit_cast(bf, h);
}
__device__ __forceinline__ float bf2f(bf b) {
  unsigned short h = __builtin_bit_cast(unsigned short, b);
  unsigned u = ((unsigned)h) << 16;
  return __builtin_bit_cast(float, u);
}

#if USE_TDM
// 2-D TDM load: tile_d1 rows x tile_d0 elements, row stride d0_stride
// (elements), element size 1<<dsz_log bytes, global tile start -> LDS addr.
// D# bitfields per CDNA5 ISA (groups 0/1; 2/3 zero).
__device__ __forceinline__ void tdm_load_2d(unsigned lds_addr, const void* gptr,
                                            unsigned dsz_log,
                                            unsigned tile_d0, unsigned tile_d1,
                                            unsigned long long d0_stride) {
  const unsigned long long ga = (unsigned long long)(uintptr_t)gptr;
  u32x4 g0;
  g0[0] = 1u;                                            // count=1 (valid D#)
  g0[1] = lds_addr;                                      // LDS byte address
  g0[2] = (unsigned)(ga & 0xFFFFFFFFu);                  // global_addr[31:0]
  g0[3] = (unsigned)((ga >> 32) & 0x01FFFFFFu) | (2u << 30);  // [56:32]|type=2
  i32x8 g1;
  g1[0] = (int)(dsz_log << 16);                          // data_size
  g1[1] = (int)((tile_d0 & 0xFFFFu) << 16);              // tensor_dim0 lo16
  g1[2] = (int)(((tile_d0 >> 16) & 0xFFFFu) |            // tensor_dim0 hi16
                ((tile_d1 & 0xFFFFu) << 16));            // tensor_dim1 lo16
  g1[3] = (int)(((tile_d1 >> 16) & 0xFFFFu) |            // tensor_dim1 hi16
                ((tile_d0 & 0xFFFFu) << 16));            // tile_dim0
  g1[4] = (int)(tile_d1 & 0xFFFFu);                      // tile_dim1 (dim2=0)
  g1[5] = (int)(d0_stride & 0xFFFFFFFFu);                // dim0_stride lo32
  g1[6] = (int)((d0_stride >> 32) & 0xFFFFu);            // dim0_stride hi16
  g1[7] = 0;
  i32x4 z4 = {0, 0, 0, 0};
#if __clang_major__ >= 23
  i32x8 z8 = {};
  __builtin_amdgcn_tensor_load_to_lds(g0, g1, z4, z4, z8, 0);
#else
  __builtin_amdgcn_tensor_load_to_lds(g0, g1, z4, z4, 0);
#endif
}
__device__ __forceinline__ unsigned lds_off(const void* p) {
  return (unsigned)(uintptr_t)p;       // generic LDS ptr: addr[31:0] = offset
}
#endif

// ---------------------------------------------------------------------------
// Generic bf16 WMMA GEMM: C[M,N] = A[M,K] * B[K,N]
// Block = 8 waves; tile = 16(M) x 128(N); k-chunks of 32; TDM path:
// double-buffered A tile + B panel prefetch, tensorcnt wait overlapped
// behind the current chunk's WMMA (1 barrier per chunk).
// ---------------------------------------------------------------------------
template<bool A_IS_F32, bool OUT_BF16>
__global__ __launch_bounds__(256) void wmma_gemm(const void* __restrict__ Ap,
                                                 const float* __restrict__ Bp,
                                                 void* __restrict__ Cp,
                                                 int M, int N, int K) {
  const int lane = threadIdx.x;
  const int half = lane >> 4;
  const int ln   = lane & 15;
  const int nbase0 = blockIdx.x * 128;
  const int nbase  = nbase0 + threadIdx.y * 16;
  const int mbase  = blockIdx.y * 16;
  v8f acc = {};

#if USE_TDM
  __shared__ float Bs[2][32 * 128];                       // [k][n], 2x16KB
  __shared__ __align__(16) char AsRaw[2][16 * 32 * (A_IS_F32 ? 4 : 2)];
  const unsigned adsz = A_IS_F32 ? 2u : 1u;
  const size_t aelt = (size_t)mbase * K;                  // A tile row 0 index

  // ---- prologue: chunk 0 ----------------------------------------------
  if (threadIdx.y == 0) {
    if (A_IS_F32)
      tdm_load_2d(lds_off(AsRaw[0]), (const float*)Ap + aelt, adsz, 32u, 16u,
                  (unsigned long long)K);
    else
      tdm_load_2d(lds_off(AsRaw[0]), (const bf*)Ap + aelt, adsz, 32u, 16u,
                  (unsigned long long)K);
    tdm_load_2d(lds_off(Bs[0]), Bp + nbase0, 2u, 128u, 32u,
                (unsigned long long)N);
    __builtin_amdgcn_s_wait_tensorcnt(0);
  }
  __syncthreads();

  const int nch = K >> 5;
  for (int ic = 0; ic < nch; ++ic) {
    const int p = ic & 1;
    // ---- prefetch chunk ic+1 into the other buffers --------------------
    if (threadIdx.y == 0 && ic + 1 < nch) {
      const int kn = (ic + 1) * 32;
      if (A_IS_F32)
        tdm_load_2d(lds_off(AsRaw[p ^ 1]), (const float*)Ap + aelt + kn, adsz,
                    32u, 16u, (unsigned long long)K);
      else
        tdm_load_2d(lds_off(AsRaw[p ^ 1]), (const bf*)Ap + aelt + kn, adsz,
                    32u, 16u, (unsigned long long)K);
      tdm_load_2d(lds_off(Bs[p ^ 1]), Bp + (size_t)kn * N + nbase0, 2u,
                  128u, 32u, (unsigned long long)N);
    }
    // ---- fragments from buffers p, one WMMA ----------------------------
    v16bf a, b;
    if (A_IS_F32) {
      const float* ar = (const float*)AsRaw[p] + ln * 32;
#pragma unroll
      for (int j = 0; j < 8; ++j) {
        a[j]     = f2bf(ar[half * 16 + j]);
        a[8 + j] = f2bf(ar[half * 16 + 8 + j]);
      }
    } else {
      const bf* ar = (const bf*)AsRaw[p] + ln * 32;
#pragma unroll
      for (int j = 0; j < 8; ++j) {
        a[j]     = ar[half * 16 + j];
        a[8 + j] = ar[half * 16 + 8 + j];
      }
    }
    const int wn = threadIdx.y * 16 + ln;
#pragma unroll
    for (int j = 0; j < 16; ++j) b[j] = f2bf(Bs[p][(half * 16 + j) * 128 + wn]);
    acc = __builtin_amdgcn_wmma_f32_16x16x32_bf16(
        false, a, false, b, (short)0, acc, false, false);
    // ---- drain the prefetch (overlapped with the WMMA above) -----------
    if (threadIdx.y == 0) __builtin_amdgcn_s_wait_tensorcnt(0);
    __syncthreads();
  }
#else
  // ---------------- fallback: manual coalesced staging ------------------
  __shared__ bf Bsm[128 * 40];       // [n][k], row stride 40 (padded)
  __shared__ bf Asm[16 * 32];        // [m][k], row stride 32
  const int tid = threadIdx.y * 32 + threadIdx.x;
  const int sm  = tid >> 4;          // A row
  const int skp = (tid & 15) * 2;    // A k pair
  const int skr = tid >> 5;          // B k row group
  const int sn4 = (tid & 31) * 4;    // B n quad
  for (int k0 = 0; k0 < K; k0 += 32) {
    if (A_IS_F32) {
      const float* A = (const float*)Ap + (size_t)(mbase + sm) * K + k0 + skp;
      Asm[sm * 32 + skp]     = f2bf(A[0]);
      Asm[sm * 32 + skp + 1] = f2bf(A[1]);
    } else {
      const bf* A = (const bf*)Ap + (size_t)(mbase + sm) * K + k0 + skp;
      Asm[sm * 32 + skp]     = A[0];
      Asm[sm * 32 + skp + 1] = A[1];
    }
#pragma unroll
    for (int it = 0; it < 4; ++it) {
      const int k = skr + it * 8;
      const float4 v = *reinterpret_cast<const float4*>(
          Bp + (size_t)(k0 + k) * N + nbase0 + sn4);
      Bsm[(sn4 + 0) * 40 + k] = f2bf(v.x);
      Bsm[(sn4 + 1) * 40 + k] = f2bf(v.y);
      Bsm[(sn4 + 2) * 40 + k] = f2bf(v.z);
      Bsm[(sn4 + 3) * 40 + k] = f2bf(v.w);
    }
    __syncthreads();
    v16bf a, b;
    const bf* ar = &Asm[ln * 32];
#pragma unroll
    for (int j = 0; j < 8; ++j) {
      a[j]     = ar[half * 16 + j];
      a[8 + j] = ar[half * 16 + 8 + j];
    }
    const bf* br = &Bsm[(threadIdx.y * 16 + ln) * 40 + half * 16];
#pragma unroll
    for (int j = 0; j < 16; ++j) b[j] = br[j];
    acc = __builtin_amdgcn_wmma_f32_16x16x32_bf16(
        false, a, false, b, (short)0, acc, false, false);
    __syncthreads();
  }
#endif

#pragma unroll
  for (int r = 0; r < 8; ++r) {
    const size_t idx = (size_t)(mbase + r + 8 * half) * N + nbase + ln;
    if (OUT_BF16) ((bf*)Cp)[idx] = f2bf(acc[r]);
    else          ((float*)Cp)[idx] = acc[r];
  }
}

// ---------------------------------------------------------------------------
// RoPE + QKV split. qkv: [B*S, 3072] bf16 ->
//   Qb [B*S, NH*HD], Kb [B*S, NKV*HD] (rotated), Vb [B*S, NKV*HD] (copied)
// ---------------------------------------------------------------------------
__global__ __launch_bounds__(256) void rope_kernel(const bf* __restrict__ qkv,
                                                   bf* __restrict__ Qb,
                                                   bf* __restrict__ Kb,
                                                   bf* __restrict__ Vb) {
  const long t = (long)blockIdx.x * blockDim.x + threadIdx.x;
  const long total = (long)B_SZ * S_SZ * 1792;
  if (t >= total) return;
  const int  r  = (int)(t % 1792);
  const long bs = t / 1792;                 // b*S + s
  const int  s  = (int)(bs % S_SZ);
  const bf* src = qkv + bs * QKV_N;

  if (r < 1280) {
    const bool isq = r < 1024;
    const int rr = isq ? r : r - 1024;
    const int hh = rr >> 6;                 // head
    const int i  = rr & 63;                 // rotary pair index
    const int col = (isq ? 0 : NH_SZ * HD_SZ) + hh * HD_SZ + i;
    const float x1 = bf2f(src[col]);
    const float x2 = bf2f(src[col + 64]);
    const float inv = __powf(10000.0f, -(float)i * (1.0f / 64.0f));
    const float ang = (float)s * inv;
    float sn, c;
    __sincosf(ang, &sn, &c);
    bf* dst = isq ? (Qb + bs * (NH_SZ * HD_SZ) + hh * HD_SZ + i)
                  : (Kb + bs * (NKV_SZ * HD_SZ) + hh * HD_SZ + i);
    dst[0]  = f2bf(x1 * c - x2 * sn);
    dst[64] = f2bf(x2 * c + x1 * sn);
  } else {
    const int rr = r - 1280;
    Vb[bs * (NKV_SZ * HD_SZ) + rr] = src[(NH_SZ + NKV_SZ) * HD_SZ + rr];
  }
}

// ---------------------------------------------------------------------------
// Per-chunk flash-attention compute (shared by TDM and fallback paths).
// KsP: [key][dim] (stride 128), VtP: [dim][key] (stride 32).
// ---------------------------------------------------------------------------
__device__ __forceinline__ void flash_chunk(const bf* __restrict__ KsP,
                                            const bf* __restrict__ VtP,
                                            const v16bf bq[4], int kk, int q,
                                            int qbase, int half, int ln,
                                            float& m_run, float& l_run,
                                            v8f accO[8]) {
  // ---- scores^T : two 16-key fragments ---------------------------------
  v8f sc0 = {}, sc1 = {};
#pragma unroll
  for (int f = 0; f < 2; ++f) {
    const bf* kr = &KsP[(f * 16 + ln) * 128];
#pragma unroll
    for (int dc = 0; dc < 4; ++dc) {
      v16bf a;
#pragma unroll
      for (int j = 0; j < 8; ++j) {
        a[j]     = kr[dc * 32 + half * 16 + j];
        a[8 + j] = kr[dc * 32 + half * 16 + 8 + j];
      }
      if (f == 0)
        sc0 = __builtin_amdgcn_wmma_f32_16x16x32_bf16(
            false, a, false, bq[dc], (short)0, sc0, false, false);
      else
        sc1 = __builtin_amdgcn_wmma_f32_16x16x32_bf16(
            false, a, false, bq[dc], (short)0, sc1, false, false);
    }
  }

  // ---- causal mask (skipped for interior chunks) + online softmax ------
  float p0[8], p1[8];
  float cmax = -1e30f;
  if (kk + 32 <= qbase) {              // fully unmasked chunk (common case)
#pragma unroll
    for (int r = 0; r < 8; ++r) {
      p0[r] = sc0[r];
      p1[r] = sc1[r];
      cmax = fmaxf(cmax, fmaxf(p0[r], p1[r]));
    }
  } else {                             // diagonal chunk: apply causal mask
#pragma unroll
    for (int r = 0; r < 8; ++r) {
      const int key0 = kk + r + 8 * half;
      const int key1 = key0 + 16;
      p0[r] = (key0 <= q) ? sc0[r] : -1e30f;
      p1[r] = (key1 <= q) ? sc1[r] : -1e30f;
      cmax = fmaxf(cmax, fmaxf(p0[r], p1[r]));
    }
  }
  cmax = fmaxf(cmax, __shfl_xor(cmax, 16, 32));
  const float m_new = fmaxf(m_run, cmax);
  const float alpha = __expf(m_run - m_new);
  float lsum = 0.0f;
#pragma unroll
  for (int r = 0; r < 8; ++r) {
    p0[r] = __expf(p0[r] - m_new);
    p1[r] = __expf(p1[r] - m_new);
    lsum += p0[r] + p1[r];
  }
  lsum += __shfl_xor(lsum, 16, 32);
  l_run = l_run * alpha + lsum;
  m_run = m_new;
#pragma unroll
  for (int dt = 0; dt < 8; ++dt)
#pragma unroll
    for (int r = 0; r < 8; ++r) accO[dt][r] *= alpha;

  // ---- pack P^T into B-fragment (32 keys x 16 q) -----------------------
  v16bf bp;
#pragma unroll
  for (int r = 0; r < 8; ++r) {
    const float o0 = __shfl_xor(p0[r], 16, 32);
    const float o1 = __shfl_xor(p1[r], 16, 32);
    const float lo = half ? o1 : p0[r];
    const float hi = half ? p1[r] : o0;
    bp[r]     = f2bf(lo);
    bp[8 + r] = f2bf(hi);
  }

  // ---- O^T += V^T x P^T -------------------------------------------------
#pragma unroll
  for (int dt = 0; dt < 8; ++dt) {
    v16bf av;
    const bf* vr = &VtP[(dt * 16 + ln) * 32];
#pragma unroll
    for (int j = 0; j < 8; ++j) {
      av[j]     = vr[half * 16 + j];
      av[8 + j] = vr[half * 16 + 8 + j];
    }
    accO[dt] = __builtin_amdgcn_wmma_f32_16x16x32_bf16(
        false, av, false, bp, (short)0, accO[dt], false, false);
  }
}

// ---------------------------------------------------------------------------
// Causal flash attention. Block = 4 waves sharing (b, kv-head).
// TDM path: triple-buffered K/V chunk prefetch (wave 0 issues chunk i+1's
// two tensor loads, s_wait_tensorcnt(2) keeps chunk i resident), V
// transposed in LDS into a double-buffered Vt; 2 barriers per chunk.
// ---------------------------------------------------------------------------
__global__ __launch_bounds__(128) void attn_kernel(const bf* __restrict__ Q,
                                                   const bf* __restrict__ Kb,
                                                   const bf* __restrict__ Vb,
                                                   bf* __restrict__ O) {
  const int tid  = threadIdx.y * 32 + threadIdx.x;   // 0..127
  const int lane = threadIdx.x;
  const int half = lane >> 4;
  const int ln   = lane & 15;
  const int bh = blockIdx.y;
  const int b = bh / NH_SZ, h = bh % NH_SZ;
  const int kvh = h / (NH_SZ / NKV_SZ);
  const int qbase = (blockIdx.x * 4 + threadIdx.y) * 16;
  const int q = qbase + ln;                     // this lane's q column
  const int nch = blockIdx.x * 2 + 2;           // chunks (32 keys each)

  // Q^T B-fragments (4 d-chunks of 32), softmax scale folded in.
  v16bf bq[4];
  {
    const bf* qp = Q + ((size_t)(b * S_SZ + q) * NH_SZ + h) * HD_SZ;
#pragma unroll
    for (int dc = 0; dc < 4; ++dc)
#pragma unroll
      for (int j = 0; j < 16; ++j)
        bq[dc][j] = f2bf(bf2f(qp[dc * 32 + half * 16 + j]) * ATT_SCALE);
  }

  float m_run = -1e30f, l_run = 0.0f;
  v8f accO[8] = {};
  const size_t kvrow = (size_t)NKV_SZ * HD_SZ;  // row stride (elements)
  const size_t base0 = ((size_t)(b * S_SZ) * NKV_SZ + kvh) * HD_SZ;

#if USE_TDM
  __shared__ bf Ks[3][32 * 128];   // [key][dim], TDM destinations
  __shared__ bf Vs[3][32 * 128];   // [key][dim], TDM destinations
  __shared__ bf Vt[2][128 * 32];   // [dim][key], transposed in LDS

  // ---- prologue: issue chunk 0 ----------------------------------------
  if (threadIdx.y == 0) {
    tdm_load_2d(lds_off(Ks[0]), Kb + base0, 1u, 128u, 32u, kvrow);
    tdm_load_2d(lds_off(Vs[0]), Vb + base0, 1u, 128u, 32u, kvrow);
  }
  for (int ic = 0; ic < nch; ++ic) {
    const int kk = ic * 32;
    const int pk = ic % 3;
    const int pv = ic & 1;
    // ---- prefetch chunk ic+1, then ensure chunk ic is resident ---------
    if (threadIdx.y == 0) {
      if (ic + 1 < nch) {
        const size_t rn = base0 + (size_t)(kk + 32) * kvrow;
        tdm_load_2d(lds_off(Ks[(ic + 1) % 3]), Kb + rn, 1u, 128u, 32u, kvrow);
        tdm_load_2d(lds_off(Vs[(ic + 1) % 3]), Vb + rn, 1u, 128u, 32u, kvrow);
        __builtin_amdgcn_s_wait_tensorcnt(2);
      } else {
        __builtin_amdgcn_s_wait_tensorcnt(0);
      }
    }
    __syncthreads();
    // ---- transpose V in LDS: Vs[pk][key][dim] -> Vt[pv][dim][key] ------
#pragma unroll
    for (int it = 0; it < 4; ++it) {
      const int cc  = it * 128 + tid;            // 0..511
      const int key = cc >> 4;
      const int d8  = (cc & 15) * 8;
      const uint4 vv = *reinterpret_cast<const uint4*>(&Vs[pk][key * 128 + d8]);
      const bf* ve = reinterpret_cast<const bf*>(&vv);
#pragma unroll
      for (int i = 0; i < 8; ++i) Vt[pv][(d8 + i) * 32 + key] = ve[i];
    }
    __syncthreads();
    if (kk < qbase + 16)            // wave-uniform: still in causal range
      flash_chunk(Ks[pk], Vt[pv], bq, kk, q, qbase, half, ln,
                  m_run, l_run, accO);
  }
#else
  // ---------------- fallback: manual coalesced staging ------------------
  __shared__ bf Ks1[32 * 128];
  __shared__ bf Vt1[128 * 32];
  for (int ic = 0; ic < nch; ++ic) {
    const int kk = ic * 32;
#pragma unroll
    for (int it = 0; it < 4; ++it) {
      const int cc  = it * 128 + tid;
      const int key = cc >> 4;
      const int d8  = (cc & 15) * 8;
      const size_t row = base0 + (size_t)(kk + key) * kvrow;
      *reinterpret_cast<uint4*>(&Ks1[key * 128 + d8]) =
          *reinterpret_cast<const uint4*>(Kb + row + d8);
      const uint4 vv = *reinterpret_cast<const uint4*>(Vb + row + d8);
      const bf* ve = reinterpret_cast<const bf*>(&vv);
#pragma unroll
      for (int i = 0; i < 8; ++i) Vt1[(d8 + i) * 32 + key] = ve[i];
    }
    __syncthreads();
    if (kk < qbase + 16)
      flash_chunk(Ks1, Vt1, bq, kk, q, qbase, half, ln, m_run, l_run, accO);
    __syncthreads();
  }
#endif

  // ---- normalize and store O[b, q, h*128 + d] as bf16 (16B per frag) ----
  const float inv_l = 1.0f / l_run;
#pragma unroll
  for (int dt = 0; dt < 8; ++dt) {
    unsigned pack[4];
#pragma unroll
    for (int w = 0; w < 4; ++w) {
      const unsigned short lo =
          __builtin_bit_cast(unsigned short, f2bf(accO[dt][2 * w] * inv_l));
      const unsigned short hi =
          __builtin_bit_cast(unsigned short, f2bf(accO[dt][2 * w + 1] * inv_l));
      pack[w] = (unsigned)lo | ((unsigned)hi << 16);
    }
    bf* op = O + (size_t)(b * S_SZ + q) * HID_SZ + h * HD_SZ + dt * 16 + half * 8;
    uint4 v; v.x = pack[0]; v.y = pack[1]; v.z = pack[2]; v.w = pack[3];
    *reinterpret_cast<uint4*>(op) = v;
  }
}

// ---------------------------------------------------------------------------
extern "C" void kernel_launch(void* const* d_in, const int* in_sizes, int n_in,
                              void* d_out, int out_size, void* d_ws, size_t ws_size,
                              hipStream_t stream) {
  const float* hidden = (const float*)d_in[0];   // [B,S,HID] f32
  const float* Wqkv   = (const float*)d_in[1];   // [HID, 3072] f32
  const float* Wo     = (const float*)d_in[2];   // [HID, HID] f32

  char* w = (char*)d_ws;
  const size_t M = (size_t)B_SZ * S_SZ;          // 8192
  bf* qkv = (bf*)w;                                           // 48 MiB
  bf* Qb  = (bf*)(w + M * QKV_N * 2);                         // 32 MiB
  bf* Kb  = (bf*)(w + M * QKV_N * 2 + M * HID_SZ * 2);        //  8 MiB
  bf* Vb  = (bf*)((char*)Kb + M * NKV_SZ * HD_SZ * 2);        //  8 MiB
  bf* Ob  = (bf*)((char*)Vb + M * NKV_SZ * HD_SZ * 2);        // 32 MiB

  dim3 gblk(32, 8);
  // 1) QKV projection: [8192,2048] x [2048,3072] -> bf16
  wmma_gemm<true, true><<<dim3(QKV_N / 128, M / 16), gblk, 0, stream>>>(
      hidden, Wqkv, qkv, (int)M, QKV_N, HID_SZ);

  // 2) RoPE + split
  const long rope_items = (long)B_SZ * S_SZ * 1792;
  rope_kernel<<<(unsigned)((rope_items + 255) / 256), 256, 0, stream>>>(
      qkv, Qb, Kb, Vb);

  // 3) Causal flash attention
  attn_kernel<<<dim3(S_SZ / 64, B_SZ * NH_SZ), dim3(32, 4), 0, stream>>>(
      Qb, Kb, Vb, Ob);

  // 4) Output projection: [8192,2048] x [2048,2048] -> f32
  wmma_gemm<false, false><<<dim3(HID_SZ / 128, M / 16), gblk, 0, stream>>>(
      Ob, Wo, d_out, (int)M, HID_SZ, HID_SZ);
}
